// DGCNN_28518582846358
// MI455X (gfx1250) — compile-verified, matrix operand order checked by reading, not run on hardware
//
#include <hip/hip_runtime.h>
#include <hip/hip_bf16.h>

typedef __attribute__((ext_vector_type(16))) _Float16 v16h;
typedef __attribute__((ext_vector_type(8)))  _Float16 v8h;
typedef __attribute__((ext_vector_type(8)))  float    v8f;

#define OUTC 226   // 128 + 1 + 32 + 32 + 32 + 1
#define FNODE 128
#define KPAD1 160  // 129 padded up to multiple of 32

#define CAT16(lo, hi) __builtin_shufflevector((lo), (hi), \
    0, 1, 2, 3, 4, 5, 6, 7, 8, 9, 10, 11, 12, 13, 14, 15)

// ---------------------------------------------------------------------------
// CSR build: row is sorted, so row_ptr[i] = lower_bound(row, i). N+1 threads.
// ---------------------------------------------------------------------------
__global__ __launch_bounds__(256) void build_row_ptr(const int* __restrict__ row,
                                                     int* __restrict__ row_ptr,
                                                     int n, int e) {
    int i = blockIdx.x * blockDim.x + threadIdx.x;
    if (i > n) return;
    int lo = 0, hi = e;
    while (lo < hi) {
        int mid = (lo + hi) >> 1;
        if (row[mid] < i) lo = mid + 1; else hi = mid;
    }
    row_ptr[i] = lo;
}

// ---------------------------------------------------------------------------
// Per-node scalars: edge_attr segment sum, degree (+ self loop), deg_inv,
// rsqrt(deg); writes out col 128 and pads the f16 layer-1 input row.
// ---------------------------------------------------------------------------
__global__ __launch_bounds__(256) void node_scalars(const float* __restrict__ ea,
                                                    const int* __restrict__ row_ptr,
                                                    float* __restrict__ deg_inv,
                                                    float* __restrict__ dsq,
                                                    float* __restrict__ out,
                                                    _Float16* __restrict__ h16,
                                                    int n) {
    int i = blockIdx.x * blockDim.x + threadIdx.x;
    if (i >= n) return;
    int e0 = row_ptr[i], e1 = row_ptr[i + 1];
    float xe = 0.f;
    for (int e = e0; e < e1; ++e) xe += ea[e];
    float deg = (float)(e1 - e0) + 1.0f;
    deg_inv[i] = 1.0f / deg;
    dsq[i]     = rsqrtf(deg);
    out[(size_t)i * OUTC + FNODE] = xe;
    _Float16* hr = h16 + (size_t)i * KPAD1;
    hr[FNODE] = (_Float16)xe;
    #pragma unroll
    for (int k = FNODE + 1; k < KPAD1; ++k) hr[k] = (_Float16)0.f;
}

// ---------------------------------------------------------------------------
// Copy x into output cols 0..127 and into the f16 layer-1 GEMM input.
// ---------------------------------------------------------------------------
__global__ __launch_bounds__(256) void copy_x(const float* __restrict__ x,
                                              float* __restrict__ out,
                                              _Float16* __restrict__ h16,
                                              int n) {
    long long t = (long long)blockIdx.x * blockDim.x + threadIdx.x;
    if (t >= (long long)n * FNODE) return;
    int i = (int)(t >> 7);
    int f = (int)(t & (FNODE - 1));
    float v = x[(size_t)i * FNODE + f];
    out[(size_t)i * OUTC + f]  = v;
    h16[(size_t)i * KPAD1 + f] = (_Float16)v;
}

// ---------------------------------------------------------------------------
// Convert weights [din, dout] f32 -> f16 pre-swizzled into the per-lane WMMA
// B-fragment layout. Linear element t decomposes as
//   t = ((kb*2 + tile)*32 + lane)*16 + j
// so the GEMM reads each fragment as two contiguous 16B vector loads/lane.
// 16-bit B-matrix layout: lane<16 -> N=lane,   K {0..7, 16..23} of the k-block;
//                         lane>=16 -> N=lane-16, K {8..15, 24..31}.
// ---------------------------------------------------------------------------
__global__ __launch_bounds__(256) void prep_W(const float* __restrict__ W,
                                              int din, int dout, int kpad,
                                              _Float16* __restrict__ Bsw) {
    int t = blockIdx.x * blockDim.x + threadIdx.x;
    if (t >= kpad * 32) return;
    int j    = t & 15;
    int lane = (t >> 4) & 31;
    int tile = (t >> 9) & 1;
    int kb   = t >> 10;
    int koff = (lane >> 4) << 3;
    int ncol = tile * 16 + (lane & 15);
    int k    = kb * 32 + ((j < 8) ? (koff + j) : (16 + koff + (j - 8)));
    float v  = (k < din && ncol < dout) ? W[(size_t)k * dout + ncol] : 0.f;
    Bsw[t] = (_Float16)v;
}

// ---------------------------------------------------------------------------
// WMMA GEMM: A [nrows, LDA] f16 row-major x pre-swizzled B -> C [nrows,32] f32.
// One wave per 16-row strip, two 16x16 output tiles, K fully unrolled
// (K=160 -> 10 WMMAs, K=32 -> 2 WMMAs). All operand loads are 16B vectors.
// ---------------------------------------------------------------------------
template <int K, int LDA>
__global__ __launch_bounds__(256) void gemm16x32(const _Float16* __restrict__ A,
                                                 const _Float16* __restrict__ Bsw,
                                                 float* __restrict__ C,
                                                 int nrows) {
    int wave  = (blockIdx.x * blockDim.x + threadIdx.x) >> 5;
    int lane  = threadIdx.x & 31;
    int tiles = nrows >> 4;
    if (wave >= tiles) return;                 // wave-uniform: EXEC stays all-1s
    int m0   = wave << 4;
    int mrow = lane & 15;
    int koff = (lane >> 4) << 3;               // 0 or 8
    int ncol = lane & 15;

    v8f c0 = {}; v8f c1 = {};
    const _Float16* arow = A + (size_t)(m0 + mrow) * LDA;

    #pragma unroll
    for (int kb = 0; kb < K / 32; ++kb) {
        v8h alo = *(const v8h*)(arow + kb * 32 + koff);
        v8h ahi = *(const v8h*)(arow + kb * 32 + 16 + koff);
        v16h a  = CAT16(alo, ahi);

        const v8h* b0p = (const v8h*)(Bsw + ((size_t)(kb * 2 + 0) * 32 + lane) * 16);
        const v8h* b1p = (const v8h*)(Bsw + ((size_t)(kb * 2 + 1) * 32 + lane) * 16);
        v16h b0 = CAT16(b0p[0], b0p[1]);
        v16h b1 = CAT16(b1p[0], b1p[1]);

        c0 = __builtin_amdgcn_wmma_f32_16x16x32_f16(false, a, false, b0,
                                                    (short)0, c0, false, false);
        c1 = __builtin_amdgcn_wmma_f32_16x16x32_f16(false, a, false, b1,
                                                    (short)0, c1, false, false);
    }
    // C/D layout: lane<16 -> (M=r, N=lane); lane>=16 -> (M=8+r, N=lane-16)
    int mb = (lane >> 4) << 3;
    #pragma unroll
    for (int r = 0; r < 8; ++r) {
        C[(size_t)(m0 + mb + r) * 32 + ncol]      = c0[r];
        C[(size_t)(m0 + mb + r) * 32 + 16 + ncol] = c1[r];
    }
}

// ---------------------------------------------------------------------------
// Edge aggregation + finalize, dout=32: one wave per node, lane = channel.
// Indices and dsq are batch-loaded 32 edges at a time (coalesced) and
// broadcast via __shfl; hw gathers are coalesced 128B lines, L2-resident.
// Deterministic: fixed per-node summation order.
// ---------------------------------------------------------------------------
__global__ __launch_bounds__(256) void agg_fin32(const float* __restrict__ hw,
                                                 const int* __restrict__ colIdx,
                                                 const int* __restrict__ row_ptr,
                                                 const float* __restrict__ dsq,
                                                 const float* __restrict__ deg_inv,
                                                 const float* __restrict__ b,
                                                 float* __restrict__ out, int col_base,
                                                 _Float16* __restrict__ h16_next,
                                                 int n) {
    int node = (blockIdx.x * blockDim.x + threadIdx.x) >> 5;
    if (node >= n) return;
    int lane = threadIdx.x & 31;
    int e0 = row_ptr[node], e1 = row_ptr[node + 1];
    float acc = 0.f;
    for (int base = e0; base < e1; base += 32) {
        int nb  = e1 - base;  if (nb > 32) nb = 32;
        int ce  = (lane < nb) ? colIdx[base + lane] : 0;
        float we = (lane < nb) ? dsq[ce] : 0.f;
        for (int j = 0; j < nb; ++j) {
            int   c = __shfl(ce, j, 32);
            float w = __shfl(we, j, 32);
            acc += hw[(size_t)c * 32 + lane] * w;
        }
    }
    float o = acc * dsq[node] + hw[(size_t)node * 32 + lane] * deg_inv[node] + b[lane];
    out[(size_t)node * OUTC + col_base + lane] = o;
    h16_next[(size_t)node * 32 + lane] = (_Float16)o;
}

// Same, dout = 1 (layer 4): thread per node, only channel 0 of hw is live.
__global__ __launch_bounds__(256) void agg_fin_d1(const float* __restrict__ hw,
                                                  const int* __restrict__ colIdx,
                                                  const int* __restrict__ row_ptr,
                                                  const float* __restrict__ dsq,
                                                  const float* __restrict__ deg_inv,
                                                  const float* __restrict__ b4,
                                                  float* __restrict__ out, int n) {
    int i = blockIdx.x * blockDim.x + threadIdx.x;
    if (i >= n) return;
    int e0 = row_ptr[i], e1 = row_ptr[i + 1];
    float acc = 0.f;
    for (int e = e0; e < e1; ++e) {
        int c = colIdx[e];
        acc += hw[(size_t)c * 32] * dsq[c];
    }
    out[(size_t)i * OUTC + 225] = acc * dsq[i] + hw[(size_t)i * 32] * deg_inv[i] + b4[0];
}

// ---------------------------------------------------------------------------
static inline char* carve(char*& p, size_t bytes) {
    char* r = p;
    p += (bytes + 255) & ~(size_t)255;
    return r;
}

extern "C" void kernel_launch(void* const* d_in, const int* in_sizes, int n_in,
                              void* d_out, int out_size, void* d_ws, size_t ws_size,
                              hipStream_t stream) {
    const float* x   = (const float*)d_in[0];
    const float* ea  = (const float*)d_in[1];
    const int*   row = (const int*)d_in[2];
    const int*   col = (const int*)d_in[3];
    const float* W1  = (const float*)d_in[4];
    const float* b1  = (const float*)d_in[5];
    const float* W2  = (const float*)d_in[6];
    const float* b2  = (const float*)d_in[7];
    const float* W3  = (const float*)d_in[8];
    const float* b3  = (const float*)d_in[9];
    const float* W4  = (const float*)d_in[10];
    const float* b4  = (const float*)d_in[11];
    float* out = (float*)d_out;

    const int N = in_sizes[0] / FNODE;   // 100000 (multiple of 16)
    const int E = in_sizes[2];           // 3200000

    char* p = (char*)d_ws;
    int*      row_ptr = (int*)     carve(p, (size_t)(N + 1) * sizeof(int));
    float*    deg_inv = (float*)   carve(p, (size_t)N * sizeof(float));
    float*    dsq     = (float*)   carve(p, (size_t)N * sizeof(float));
    _Float16* h16a    = (_Float16*)carve(p, (size_t)N * KPAD1 * sizeof(_Float16));
    _Float16* h16b    = (_Float16*)carve(p, (size_t)N * 32 * sizeof(_Float16));
    float*    hw      = (float*)   carve(p, (size_t)N * 32 * sizeof(float));
    _Float16* Bsw     = (_Float16*)carve(p, (size_t)KPAD1 * 32 * sizeof(_Float16));

    const int T = 256;
    const int gemm_blocks = ((N / 16) * 32 + T - 1) / T;
    const int wavekernel_blocks = (int)(((long long)N * 32 + T - 1) / T);

    build_row_ptr<<<(N + 1 + T - 1) / T, T, 0, stream>>>(row, row_ptr, N, E);
    node_scalars<<<(N + T - 1) / T, T, 0, stream>>>(ea, row_ptr, deg_inv, dsq, out, h16a, N);
    copy_x<<<(int)(((long long)N * FNODE + T - 1) / T), T, 0, stream>>>(x, out, h16a, N);

    // Layer 1: [N,129(pad160)] @ [129,32]
    prep_W<<<(KPAD1 * 32 + T - 1) / T, T, 0, stream>>>(W1, 129, 32, KPAD1, Bsw);
    gemm16x32<KPAD1, KPAD1><<<gemm_blocks, T, 0, stream>>>(h16a, Bsw, hw, N);
    agg_fin32<<<wavekernel_blocks, T, 0, stream>>>(hw, col, row_ptr, dsq, deg_inv, b1,
                                                   out, 129, h16b, N);
    // Layer 2: [N,32] @ [32,32]
    prep_W<<<(32 * 32 + T - 1) / T, T, 0, stream>>>(W2, 32, 32, 32, Bsw);
    gemm16x32<32, 32><<<gemm_blocks, T, 0, stream>>>(h16b, Bsw, hw, N);
    agg_fin32<<<wavekernel_blocks, T, 0, stream>>>(hw, col, row_ptr, dsq, deg_inv, b2,
                                                   out, 161, h16b, N);
    // Layer 3: [N,32] @ [32,32]
    prep_W<<<(32 * 32 + T - 1) / T, T, 0, stream>>>(W3, 32, 32, 32, Bsw);
    gemm16x32<32, 32><<<gemm_blocks, T, 0, stream>>>(h16b, Bsw, hw, N);
    agg_fin32<<<wavekernel_blocks, T, 0, stream>>>(hw, col, row_ptr, dsq, deg_inv, b3,
                                                   out, 193, h16b, N);
    // Layer 4: [N,32] @ [32,1] (weights zero-padded to 32 cols)
    prep_W<<<(32 * 32 + T - 1) / T, T, 0, stream>>>(W4, 32, 1, 32, Bsw);
    gemm16x32<32, 32><<<gemm_blocks, T, 0, stream>>>(h16b, Bsw, hw, N);
    agg_fin_d1<<<(N + T - 1) / T, T, 0, stream>>>(hw, col, row_ptr, dsq, deg_inv, b4, out, N);
}
